// CamLiPWC_Core_90323162235025
// MI455X (gfx1250) — matrix operand order repaired and checked by reading, not compiled, not run on hardware
//
#include <hip/hip_runtime.h>

typedef __attribute__((ext_vector_type(8)))  _Float16 v8h;
typedef __attribute__((ext_vector_type(16))) _Float16 v16h;
typedef __attribute__((ext_vector_type(8)))  float    v8f;

namespace {
constexpr int B_  = 2;
constexpr int H_  = 48;
constexpr int W_  = 160;
constexpr int N_  = 4096;
constexpr int C_  = 64;
constexpr int K_  = 16;
constexpr int MD_ = 4;
constexpr int HW_ = H_ * W_;       // 7680
constexpr int R2_ = B_ * HW_;      // 15360 rows (2D)  (div by 32)
constexpr int R3_ = B_ * N_;       // 8192 rows (3D)   (div by 32)
constexpr int RKr_ = R3_ * K_;     // 131072 rows      (div by 32)
constexpr int CORR_ = 81;
}

struct SegMap { int n; int src[8]; int dst[8]; int len[8]; };

// ---------------------------------------------------------------------------
// Weight fp32 [Co,Ci,taps] -> f16 [CoPad, taps*Kpad] with channel remapping.
// Destination buffer is pre-zeroed, so pads contribute exact zeros to GEMM.
// ---------------------------------------------------------------------------
__global__ void k_wcvt(const float* __restrict__ w, _Float16* __restrict__ o,
                       int Co, int Ci, int taps, int Kpad, SegMap sm)
{
    long long i = (long long)blockIdx.x * blockDim.x + threadIdx.x;
    long long tot = (long long)Co * Ci * taps;
    if (i >= tot) return;
    int t = (int)(i % taps);
    long long r = i / taps;
    int ci = (int)(r % Ci);
    int co = (int)(r / Ci);
    int dst = -1;
    for (int s = 0; s < sm.n; ++s) {
        int d = ci - sm.src[s];
        if (d >= 0 && d < sm.len[s]) { dst = sm.dst[s] + d; break; }
    }
    if (dst < 0) return;
    o[((long long)co * taps + t) * Kpad + dst] = (_Float16)w[i];
}

__global__ void k_copyf(const float* __restrict__ a, float* __restrict__ b, int n)
{
    int i = blockIdx.x * blockDim.x + threadIdx.x;
    if (i < n) b[i] = a[i];
}

// NCHW fp32 -> rows [b*spatial, stride] f16 at channel offset.
__global__ void k_rows_from_nchw(const float* __restrict__ in, _Float16* __restrict__ out,
                                 int Cin, int spatial, int stride, int chOff, int batches)
{
    long long i = (long long)blockIdx.x * blockDim.x + threadIdx.x;
    long long tot = (long long)batches * spatial * Cin;
    if (i >= tot) return;
    int c = (int)(i % Cin);
    long long r = i / Cin;
    int b = (int)(r / spatial);
    int p = (int)(r - (long long)b * spatial);
    out[r * stride + chOff + c] = (_Float16)in[((long long)b * Cin + c) * spatial + p];
}

__device__ __forceinline__ int clampi(int x, int lo, int hi) { return x < lo ? lo : (x > hi ? hi : x); }
__device__ __forceinline__ float leakyf(float v) { return v > 0.f ? v : 0.1f * v; }

__device__ __forceinline__ v16h ldA(const _Float16* p)
{
    // dense 16-bit A layout: elems 0..7 = K..K+7, elems 8..15 = K+16..K+23
    union { v16h v; v8h h[2]; } u;
    u.h[0] = *(const v8h*)(p);
    u.h[1] = *(const v8h*)(p + 16);
    return u.v;
}
__device__ __forceinline__ v16h ldB(const _Float16* p)
{
    // B layout: elems 0..7 = K..K+7, elems 8..15 = K+16..K+23 (matches A)
    union { v16h v; v8h h[2]; } u;
    u.h[0] = *(const v8h*)(p);
    u.h[1] = *(const v8h*)(p + 16);
    return u.v;
}

// ---------------------------------------------------------------------------
// Core WMMA implicit-GEMM conv / pointwise GEMM.
// One wave computes a 32(M=out-ch) x 32(N=row) block = 2x2 WMMA tiles:
// 4 v_wmma per K-step reusing 2 A + 2 B operand fetches.
// taps==1: plain GEMM over rows. taps==9: 3x3 conv (dilated) over NHWC rows,
// literal-unrolled tap loop, halo rows redirected to a zeroed scratch row.
// ---------------------------------------------------------------------------
__global__ void __launch_bounds__(256)
k_wmma_conv(const _Float16* __restrict__ inRows,
            const _Float16* __restrict__ wgt,
            const float* __restrict__ bias,
            const _Float16* __restrict__ zrow,
            _Float16* __restrict__ outRows, float* __restrict__ outF32,
            int rows, int Kpad, int taps, int dil,
            int Hh, int Wv, int inStride,
            int CoPad, int CoReal, int outStride, int outChOff,
            int o32Stride, int act)
{
    const int wid  = (int)((blockIdx.x * (unsigned)blockDim.x + threadIdx.x) >> 5);
    const int lane = threadIdx.x & 31;
    const int tilesM = CoPad >> 5;
    const int tilesN = rows >> 5;
    if (wid >= tilesM * tilesN) return;
    const int m0 = (wid % tilesM) << 5;
    const int n0 = (wid / tilesM) << 5;
    const int l  = lane & 15;
    const int hi = lane >> 4;
    const int nrow0 = n0 + l;
    const int nrow1 = n0 + 16 + l;

    const long long wstride = (long long)taps * Kpad;
    const _Float16* wrow0 = wgt + (long long)(m0 + l) * wstride;
    const _Float16* wrow1 = wrow0 + 16 * wstride;

    const int kbA = hi * 8;    // A: lane-half selects K sub-block
    const int kbB = hi * 8;    // B: lane-half selects K sub-block

    v8f acc00 = {}, acc01 = {}, acc10 = {}, acc11 = {};

    if (taps == 1) {
        const _Float16* b0 = inRows + (long long)nrow0 * inStride + kbB;
        const _Float16* b1 = inRows + (long long)nrow1 * inStride + kbB;
        const _Float16* w0 = wrow0 + kbA;
        const _Float16* w1 = wrow1 + kbA;
        for (int k0 = 0; k0 < Kpad; k0 += 32) {
            v16h a0 = ldA(w0 + k0), a1 = ldA(w1 + k0);
            v16h vb0 = ldB(b0 + k0), vb1 = ldB(b1 + k0);
            acc00 = __builtin_amdgcn_wmma_f32_16x16x32_f16(false, a0, false, vb0, (short)0, acc00, false, false);
            acc01 = __builtin_amdgcn_wmma_f32_16x16x32_f16(false, a0, false, vb1, (short)0, acc01, false, false);
            acc10 = __builtin_amdgcn_wmma_f32_16x16x32_f16(false, a1, false, vb0, (short)0, acc10, false, false);
            acc11 = __builtin_amdgcn_wmma_f32_16x16x32_f16(false, a1, false, vb1, (short)0, acc11, false, false);
        }
    } else {
        const int sp = Hh * Wv;
        const int bb0 = nrow0 / sp;
        const int p0  = nrow0 - bb0 * sp;
        const int yy0 = p0 / Wv, xx0 = p0 - (p0 / Wv) * Wv;
        const int bb1 = nrow1 / sp;
        const int p1  = nrow1 - bb1 * sp;
        const int yy1 = p1 / Wv, xx1 = p1 - (p1 / Wv) * Wv;
#pragma unroll
        for (int t = 0; t < 9; ++t) {
            const int dy = (t / 3 - 1) * dil;
            const int dx = (t % 3 - 1) * dil;
            const int ya = yy0 + dy, xa = xx0 + dx;
            const int yb = yy1 + dy, xb = xx1 + dx;
            const bool oka = (ya >= 0) & (ya < Hh) & (xa >= 0) & (xa < Wv);
            const bool okb = (yb >= 0) & (yb < Hh) & (xb >= 0) & (xb < Wv);
            const _Float16* b0 = (oka ? inRows + (long long)((bb0 * Hh + ya) * Wv + xa) * inStride : zrow) + kbB;
            const _Float16* b1 = (okb ? inRows + (long long)((bb1 * Hh + yb) * Wv + xb) * inStride : zrow) + kbB;
            __builtin_prefetch((const void*)b0, 0, 0);
            __builtin_prefetch((const void*)b1, 0, 0);
            const _Float16* w0 = wrow0 + t * Kpad + kbA;
            const _Float16* w1 = wrow1 + t * Kpad + kbA;
            for (int k0 = 0; k0 < Kpad; k0 += 32) {
                v16h a0 = ldA(w0 + k0), a1 = ldA(w1 + k0);
                v16h vb0 = ldB(b0 + k0), vb1 = ldB(b1 + k0);
                acc00 = __builtin_amdgcn_wmma_f32_16x16x32_f16(false, a0, false, vb0, (short)0, acc00, false, false);
                acc01 = __builtin_amdgcn_wmma_f32_16x16x32_f16(false, a0, false, vb1, (short)0, acc01, false, false);
                acc10 = __builtin_amdgcn_wmma_f32_16x16x32_f16(false, a1, false, vb0, (short)0, acc10, false, false);
                acc11 = __builtin_amdgcn_wmma_f32_16x16x32_f16(false, a1, false, vb1, (short)0, acc11, false, false);
            }
        }
    }

    // D layout: VGPR e -> M = mBase + hi*8 + e, N = n0(+16) + (lane&15)
#pragma unroll
    for (int q = 0; q < 4; ++q) {
        const v8f& a = (q == 0) ? acc00 : (q == 1) ? acc01 : (q == 2) ? acc10 : acc11;
        const int mb   = m0 + ((q >> 1) ? 16 : 0) + hi * 8;
        const int orow = ((q & 1) ? nrow1 : nrow0);
#pragma unroll
        for (int e = 0; e < 8; ++e) {
            const int ch = mb + e;
            float v = a[e] + bias[ch];
            if (act == 1) v = leakyf(v);
            else if (act == 2) v = fmaxf(v, 0.f);
            if (ch < CoReal) {
                if (outRows) outRows[(long long)orow * outStride + outChOff + ch] = (_Float16)v;
                if (outF32)  outF32[(long long)orow * o32Stride + ch] = v;
            }
        }
    }
}

// ---------------------------------------------------------------------------
// Irregular-op kernels
// ---------------------------------------------------------------------------
__global__ void k_project(const float* __restrict__ xyz, const float* __restrict__ cam,
                          float* __restrict__ uv, float* __restrict__ mask)
{
    int i = blockIdx.x * blockDim.x + threadIdx.x;
    if (i >= R3_) return;
    int b = i / N_, n = i - b * N_;
    float x = xyz[((long long)b * 3 + 0) * N_ + n];
    float y = xyz[((long long)b * 3 + 1) * N_ + n];
    float z = xyz[((long long)b * 3 + 2) * N_ + n];
    float zs = fmaxf(z, 1e-6f);
    float u = cam[0] * x / zs + cam[2];
    float v = cam[1] * y / zs + cam[3];
    float m = (u >= 0.f && u <= (float)(W_ - 1) && v >= 0.f && v <= (float)(H_ - 1) && z > 0.f) ? 1.f : 0.f;
    uv[2 * i] = u; uv[2 * i + 1] = v; mask[i] = m;
}

__global__ void k_knn(const float* __restrict__ q, const float* __restrict__ qf,
                      const float* __restrict__ ref, int* __restrict__ idx)
{
    int i = blockIdx.x * blockDim.x + threadIdx.x;
    if (i >= R3_) return;
    int b = i / N_, n = i - b * N_;
    const float* qb = q + (long long)b * 3 * N_;
    const float* rb = ref + (long long)b * 3 * N_;
    float qx = qb[n], qy = qb[N_ + n], qz = qb[2 * N_ + n];
    if (qf) {
        qx += qf[((long long)b * 3 + 0) * N_ + n];
        qy += qf[((long long)b * 3 + 1) * N_ + n];
        qz += qf[((long long)b * 3 + 2) * N_ + n];
    }
    float bd[K_]; int bi[K_];
#pragma unroll
    for (int j = 0; j < K_; ++j) { bd[j] = 3.4e38f; bi[j] = 0; }
    for (int m = 0; m < N_; ++m) {
        float dx = qx - rb[m], dy = qy - rb[N_ + m], dz = qz - rb[2 * N_ + m];
        float d = dx * dx + dy * dy + dz * dz;
        if (d < bd[K_ - 1]) {
            float cd = d; int ci = m;
#pragma unroll
            for (int j = 0; j < K_; ++j) {
                if (cd < bd[j]) {
                    float td = bd[j]; int ti = bi[j];
                    bd[j] = cd; bi[j] = ci; cd = td; ci = ti;
                }
            }
        }
    }
#pragma unroll
    for (int j = 0; j < K_; ++j) idx[(long long)i * K_ + j] = bi[j];
}

__global__ void k_gather(const _Float16* __restrict__ src, int srcStride, int srcChOff, int C,
                         const float* __restrict__ uv, const float* __restrict__ mask,
                         _Float16* __restrict__ dst, int dstStride, int dstChOff)
{
    long long i = (long long)blockIdx.x * blockDim.x + threadIdx.x;
    long long tot = (long long)R3_ * C;
    if (i >= tot) return;
    int c = (int)(i % C);
    int r = (int)(i / C);
    int b = r / N_;
    int ui = clampi((int)rintf(uv[2 * r]), 0, W_ - 1);
    int vi = clampi((int)rintf(uv[2 * r + 1]), 0, H_ - 1);
    long long g = (long long)(b * H_ + vi) * W_ + ui;
    float v = (float)src[g * srcStride + srcChOff + c] * mask[r];
    dst[(long long)r * dstStride + dstChOff + c] = (_Float16)v;
}

__global__ void k_scatter(const _Float16* __restrict__ f3d, int stride, int chOff, int C,
                          const float* __restrict__ uv, const float* __restrict__ mask,
                          float* __restrict__ acc, float* __restrict__ cnt)
{
    long long i = (long long)blockIdx.x * blockDim.x + threadIdx.x;
    long long tot = (long long)R3_ * C;
    if (i >= tot) return;
    int c = (int)(i % C);
    int r = (int)(i / C);
    int b = r / N_;
    int ui = clampi((int)rintf(uv[2 * r]), 0, W_ - 1);
    int vi = clampi((int)rintf(uv[2 * r + 1]), 0, H_ - 1);
    long long g = (long long)b * HW_ + vi * W_ + ui;
    float m = mask[r];
    atomicAdd(&acc[g * C + c], (float)f3d[(long long)r * stride + chOff + c] * m);
    if (c == 0) atomicAdd(&cnt[g], m);
}

__global__ void k_scatnorm(const float* __restrict__ acc, const float* __restrict__ cnt, int C,
                           _Float16* __restrict__ dst, int stride, int chOff)
{
    long long i = (long long)blockIdx.x * blockDim.x + threadIdx.x;
    long long tot = (long long)R2_ * C;
    if (i >= tot) return;
    int c = (int)(i % C);
    long long g = i / C;
    dst[g * stride + chOff + c] = (_Float16)(acc[g * C + c] / fmaxf(cnt[g], 1.f));
}

__global__ void k_backwarp(const _Float16* __restrict__ feat, int stride,
                           const float* __restrict__ flow,   // NCHW (B,2,HW)
                           _Float16* __restrict__ out, int outStride)
{
    long long i = (long long)blockIdx.x * blockDim.x + threadIdx.x;
    long long tot = (long long)R2_ * C_;
    if (i >= tot) return;
    int c = (int)(i % C_);
    int r = (int)(i / C_);
    int b = r / HW_, p = r - b * HW_;
    int py = p / W_, px = p - py * W_;
    float x = px + flow[((long long)b * 2 + 0) * HW_ + p];
    float y = py + flow[((long long)b * 2 + 1) * HW_ + p];
    float x0 = floorf(x), y0 = floorf(y);
    float wx = x - x0, wy = y - y0;
    int xi0 = clampi((int)x0, 0, W_ - 1), xi1 = clampi((int)x0 + 1, 0, W_ - 1);
    int yi0 = clampi((int)y0, 0, H_ - 1), yi1 = clampi((int)y0 + 1, 0, H_ - 1);
    long long base = (long long)b * HW_;
    float v00 = (float)feat[(base + yi0 * W_ + xi0) * stride + c];
    float v01 = (float)feat[(base + yi0 * W_ + xi1) * stride + c];
    float v10 = (float)feat[(base + yi1 * W_ + xi0) * stride + c];
    float v11 = (float)feat[(base + yi1 * W_ + xi1) * stride + c];
    float v = v00 * (1.f - wx) * (1.f - wy) + v01 * wx * (1.f - wy)
            + v10 * (1.f - wx) * wy + v11 * wx * wy;
    out[(long long)r * outStride + c] = (_Float16)v;
}

__global__ void k_corr(const _Float16* __restrict__ f1, int s1,
                       const _Float16* __restrict__ f2, int s2,
                       _Float16* __restrict__ out, int so)
{
    long long i = (long long)blockIdx.x * blockDim.x + threadIdx.x;
    long long tot = (long long)R2_ * CORR_;
    if (i >= tot) return;
    int o = (int)(i % CORR_);
    int r = (int)(i / CORR_);
    int b = r / HW_, p = r - b * HW_;
    int py = p / W_, px = p - py * W_;
    int y2 = py + o / 9 - MD_;
    int x2 = px + o % 9 - MD_;
    float s = 0.f;
    if (y2 >= 0 && y2 < H_ && x2 >= 0 && x2 < W_) {
        const _Float16* a = f1 + (long long)r * s1;
        const _Float16* c2 = f2 + (long long)((b * H_ + y2) * W_ + x2) * s2;
        for (int c = 0; c < C_; ++c) s += (float)a[c] * (float)c2[c];
    }
    out[(long long)r * so + o] = (_Float16)leakyf(s * (1.f / 64.f));
}

// cat1 = [f1_3d(bcast over k) 64 | f2_3d[idx2] 64 | xyz2[idx2]-q 3] -> [RK,160]
__global__ void k_cat1(const _Float16* __restrict__ f1, const _Float16* __restrict__ f2,
                       const float* __restrict__ xyz1, const float* __restrict__ fl3,
                       const float* __restrict__ xyz2, const int* __restrict__ idx2,
                       _Float16* __restrict__ out)
{
    long long i = (long long)blockIdx.x * blockDim.x + threadIdx.x;
    long long tot = (long long)RKr_ * 131;
    if (i >= tot) return;
    int c = (int)(i % 131);
    long long rk = i / 131;
    long long r = rk / K_;
    int b = (int)(r / N_), n = (int)(r - (long long)b * N_);
    int j = idx2[rk];
    _Float16 v;
    if (c < 64) v = f1[r * 64 + c];
    else if (c < 128) v = f2[((long long)b * N_ + j) * 64 + (c - 64)];
    else {
        int d = c - 128;
        float a = xyz2[((long long)b * 3 + d) * N_ + j];
        float qq = xyz1[((long long)b * 3 + d) * N_ + n] + fl3[((long long)b * 3 + d) * N_ + n];
        v = (_Float16)(a - qq);
    }
    out[rk * 160 + c] = v;
}

// cat2 = [hmax[idx1] 64 | xyz1[idx1]-xyz1 3] -> [RK,96]
__global__ void k_cat2(const _Float16* __restrict__ hmax, const float* __restrict__ xyz1,
                       const int* __restrict__ idx1, _Float16* __restrict__ out)
{
    long long i = (long long)blockIdx.x * blockDim.x + threadIdx.x;
    long long tot = (long long)RKr_ * 67;
    if (i >= tot) return;
    int c = (int)(i % 67);
    long long rk = i / 67;
    long long r = rk / K_;
    int b = (int)(r / N_), n = (int)(r - (long long)b * N_);
    int j = idx1[rk];
    _Float16 v;
    if (c < 64) v = hmax[((long long)b * N_ + j) * 64 + c];
    else {
        int d = c - 64;
        v = (_Float16)(xyz1[((long long)b * 3 + d) * N_ + j] - xyz1[((long long)b * 3 + d) * N_ + n]);
    }
    out[rk * 96 + c] = v;
}

__global__ void k_maxk(const _Float16* __restrict__ in, int stride, int C,
                       _Float16* __restrict__ out, int ostride, int chOff)
{
    long long i = (long long)blockIdx.x * blockDim.x + threadIdx.x;
    long long tot = (long long)R3_ * C;
    if (i >= tot) return;
    int c = (int)(i % C);
    long long r = i / C;
    float m = -3.4e38f;
    for (int k = 0; k < K_; ++k) m = fmaxf(m, (float)in[(r * K_ + k) * stride + c]);
    out[r * ostride + chOff + c] = (_Float16)m;
}

__global__ void k_maxgroup(const _Float16* __restrict__ in, int stride, int C,
                           const int* __restrict__ idx, _Float16* __restrict__ out, int ostride)
{
    long long i = (long long)blockIdx.x * blockDim.x + threadIdx.x;
    long long tot = (long long)R3_ * C;
    if (i >= tot) return;
    int c = (int)(i % C);
    long long r = i / C;
    int b = (int)(r / N_);
    float m = -3.4e38f;
    for (int k = 0; k < K_; ++k) {
        int j = idx[r * K_ + k];
        m = fmaxf(m, (float)in[((long long)b * N_ + j) * stride + c]);
    }
    out[r * ostride + c] = (_Float16)m;
}

__global__ void k_flow2d_mid(const float* __restrict__ last, const float* __restrict__ delta,
                             float* __restrict__ v1, _Float16* __restrict__ hc, int hcStride, int chOff)
{
    int i = blockIdx.x * blockDim.x + threadIdx.x;
    if (i >= R2_ * 2) return;
    int c = i % 2, r = i / 2;
    int b = r / HW_, p = r - b * HW_;
    float v = last[((long long)b * 2 + c) * HW_ + p] + delta[i];
    v1[i] = v;
    hc[(long long)r * hcStride + chOff + c] = (_Float16)v;
}

__global__ void k_flow2d_final(const float* __restrict__ v1, const float* __restrict__ d2,
                               float* __restrict__ out)
{
    int i = blockIdx.x * blockDim.x + threadIdx.x;
    if (i >= R2_ * 2) return;
    int c = i % 2, r = i / 2;
    int b = r / HW_, p = r - b * HW_;
    out[((long long)b * 2 + c) * HW_ + p] = v1[i] + d2[i];
}

__global__ void k_flow3d_final(const float* __restrict__ last3, const float* __restrict__ delta,
                               float* __restrict__ out)
{
    int i = blockIdx.x * blockDim.x + threadIdx.x;
    if (i >= R3_ * 3) return;
    int c = i % 3, r = i / 3;
    int b = r / N_, n = r - b * N_;
    out[((long long)b * 3 + c) * N_ + n] = last3[((long long)b * 3 + c) * N_ + n] + delta[(long long)r * 3 + c];
}

__global__ void k_rows_to_nchw(const float* __restrict__ rows, int C, float* __restrict__ out)
{
    long long i = (long long)blockIdx.x * blockDim.x + threadIdx.x;
    long long tot = (long long)R2_ * C;
    if (i >= tot) return;
    int c = (int)(i % C);
    int r = (int)(i / C);
    int b = r / HW_, p = r - b * HW_;
    out[((long long)b * C + c) * HW_ + p] = rows[i];
}

// ---------------------------------------------------------------------------
// Host orchestration
// ---------------------------------------------------------------------------
static SegMap segId(int len)
{
    SegMap s{}; s.n = 1; s.src[0] = 0; s.dst[0] = 0; s.len[0] = len; return s;
}
static SegMap segEst(int n)
{
    static const int ss[6] = {0, 179, 307, 435, 531, 595};
    static const int dd[6] = {0, 192, 320, 448, 544, 608};
    static const int ll[6] = {179, 128, 128, 96, 64, 32};
    SegMap s{}; s.n = n;
    for (int i = 0; i < n; ++i) { s.src[i] = ss[i]; s.dst[i] = dd[i]; s.len[i] = ll[i]; }
    return s;
}
static inline dim3 g1(long long n) { return dim3((unsigned)((n + 255) / 256)); }

extern "C" void kernel_launch(void* const* d_in, const int* in_sizes, int n_in,
                              void* d_out, int out_size, void* d_ws, size_t ws_size,
                              hipStream_t stream)
{
    if (n_in < 71 || out_size < R2_ * 2 + R3_ * 3 + R2_ * 144) return;

    const float* feat1_2d = (const float*)d_in[0];
    const float* feat2_2d = (const float*)d_in[1];
    const float* feat1_3d = (const float*)d_in[2];
    const float* feat2_3d = (const float*)d_in[3];
    const float* xyz1     = (const float*)d_in[4];
    const float* xyz2     = (const float*)d_in[5];
    const float* lf2d     = (const float*)d_in[6];
    const float* lff2d    = (const float*)d_in[7];
    const float* lf3d     = (const float*)d_in[8];
    const float* lff3d    = (const float*)d_in[9];
    const float* cam      = (const float*)d_in[10];
    const float* pp[60];
    for (int i = 0; i < 60; ++i) pp[i] = (const float*)d_in[11 + i];

    float* outF     = (float*)d_out;
    float* outFlow2 = outF;
    float* outFlow3 = outF + (long long)R2_ * 2;
    float* outUm    = outF + (long long)R2_ * 2 + (long long)R3_ * 3;

    // bump allocator over d_ws (pointer math only; memset + launches after)
    char* base = (char*)d_ws;
    size_t off = 0;
    auto alloc = [&](size_t bytes) -> void* {
        void* p = base + off;
        off += (bytes + 255) & ~(size_t)255;
        return p;
    };
    auto aH = [&](long long n) { return (_Float16*)alloc((size_t)n * sizeof(_Float16)); };
    auto aF = [&](long long n) { return (float*)alloc((size_t)n * sizeof(float)); };
    auto aI = [&](long long n) { return (int*)alloc((size_t)n * sizeof(int)); };

    // ---- weights -----------------------------------------------------------
    struct WSpec { const float* w; const float* b; int Co, Ci, taps, Kpad, CoPad; SegMap sm; _Float16* wf; float* bf; };
    WSpec WS[32]; int nws = 0;
    auto addW = [&](const float* w, const float* b, int Co, int Ci, int taps, int Kpad, SegMap sm) -> int {
        WSpec q; q.w = w; q.b = b; q.Co = Co; q.Ci = Ci; q.taps = taps; q.Kpad = Kpad; q.sm = sm;
        q.CoPad = (Co + 31) & ~31;   // M-tiling is 32-wide
        q.wf = (_Float16*)alloc((size_t)q.CoPad * taps * Kpad * sizeof(_Float16));
        q.bf = (float*)alloc((size_t)q.CoPad * sizeof(float));
        WS[nws] = q; return nws++;
    };

    const int iWp2d  = addW(pp[0],  pp[1],  64, 128, 1, 128, segId(128));
    const int iWp3d  = addW(pp[2],  pp[3],  64, 128, 1, 128, segId(128));
    const int iWc1   = addW(pp[4],  pp[5],  64, 131, 1, 160, segId(131));
    const int iWc2   = addW(pp[6],  pp[7],  64,  67, 1,  96, segId(67));
    const int iWf2d  = addW(pp[8],  pp[9],  81, 145, 1, 160, segId(145));
    const int iWf3d  = addW(pp[10], pp[11], 64, 145, 1, 160, segId(145));
    const int iWa2d  = addW(pp[12], pp[13], 64,  64, 9,  64, segId(64));
    const int iWa3d  = addW(pp[14], pp[15], 64,  64, 1,  64, segId(64));
    const int iWac3d = addW(pp[16], pp[17], 64,  64, 1,  64, segId(64));
    const int estCi[5] = {179, 307, 435, 531, 595};
    const int estCo[5] = {128, 128, 96, 64, 32};
    const int estKp[5] = {192, 320, 448, 544, 608};
    int iWe[5];
    for (int i = 0; i < 5; ++i)
        iWe[i] = addW(pp[18 + 2 * i], pp[19 + 2 * i], estCo[i], estCi[i], 9, estKp[i], segEst(i + 1));
    const int iWfp1 = addW(pp[28], pp[29], 128, 195, 1, 224, segId(195));
    const int iWfp2 = addW(pp[30], pp[31], 128, 128, 1, 128, segId(128));
    const int iWfp3 = addW(pp[32], pp[33],  64, 128, 1, 128, segId(128));
    SegMap smEf2d = segEst(6);
    smEf2d.src[6] = 627; smEf2d.dst[6] = 640; smEf2d.len[6] = 64; smEf2d.n = 7;
    const int iWef2d = addW(pp[34], pp[35], 627, 691, 1, 704, smEf2d);
    SegMap smEf3d{};
    {
        const int ss[6] = {0, 243, 371, 499, 595, 659};
        const int dd[6] = {0, 256, 384, 512, 608, 672};
        const int ll[6] = {243, 128, 128, 96, 64, 32};
        smEf3d.n = 6;
        for (int i = 0; i < 6; ++i) { smEf3d.src[i] = ss[i]; smEf3d.dst[i] = dd[i]; smEf3d.len[i] = ll[i]; }
    }
    const int iWef3d = addW(pp[36], pp[37], 64, 691, 1, 704, smEf3d);
    const int iWl2d  = addW(pp[38], pp[39], 2, 627, 9, 640, segId(627));
    const int iWl3d  = addW(pp[40], pp[41], 3,  64, 1,  64, segId(64));
    const int ctxCi[6] = {629, 128, 128, 128, 96, 64};
    const int ctxCo[6] = {128, 128, 128, 96, 64, 32};
    const int ctxKp[6] = {640, 128, 128, 128, 96, 64};
    int iWctx[6];
    for (int i = 0; i < 6; ++i)
        iWctx[i] = addW(pp[42 + 2 * i], pp[43 + 2 * i], ctxCo[i], ctxCi[i], 9, ctxKp[i], segId(ctxCi[i]));
    const int iWcf  = addW(pp[54], pp[55],   2,  32, 9,  32, segId(32));
    const int iWum1 = addW(pp[56], pp[57], 256,  32, 9,  32, segId(32));
    const int iWum2 = addW(pp[58], pp[59], 144, 256, 1, 256, segId(256));

    // ---- activations -------------------------------------------------------
    _Float16* zrow   = aH(1024);                    // stays all-zero: conv halo rows
    _Float16* cat2d1 = aH((long long)R2_ * 128);    // [feat1_2d | scat1]
    _Float16* cat2d2 = aH((long long)R2_ * 128);
    _Float16* cat3d1 = aH((long long)R3_ * 128);    // [feat1_3d | samp1]
    _Float16* cat3d2 = aH((long long)R3_ * 128);
    _Float16* f1_2d  = aH((long long)R2_ * 64);
    _Float16* f2_2d  = aH((long long)R2_ * 64);
    _Float16* f1_3d  = aH((long long)R3_ * 64);
    _Float16* f2_3d  = aH((long long)R3_ * 64);
    _Float16* warp2  = aH((long long)R2_ * 64);
    _Float16* corr2c = aH((long long)R2_ * 160);    // [corr2d 81 | scat3d 64 | pad]
    _Float16* corr3c = aH((long long)R3_ * 160);    // [corr3d 64 | samp 81 | pad]
    _Float16* cat1b  = aH((long long)RKr_ * 160);
    _Float16* hbuf   = aH((long long)RKr_ * 64);
    _Float16* hmax   = aH((long long)R3_ * 64);
    _Float16* cat2b  = aH((long long)RKr_ * 96);
    _Float16* h2buf  = aH((long long)RKr_ * 64);
    _Float16* corr3f = aH((long long)R3_ * 64);
    _Float16* estB   = aH((long long)R2_ * 704);    // x2d + est chain + scat(ff3d)
    _Float16* x3dB   = aH((long long)R3_ * 224);
    _Float16* h3a    = aH((long long)R3_ * 128);
    _Float16* h3b    = aH((long long)R3_ * 128);
    _Float16* h3c    = aH((long long)R3_ * 128);
    _Float16* cat33  = aH((long long)R3_ * 704);    // [ff3d 64 | samp(estB) 640]
    _Float16* ff3dn  = aH((long long)R3_ * 64);
    _Float16* hcB    = aH((long long)R2_ * 640);    // ff2d(new) + flow2d
    const int cS[6] = {128, 128, 128, 96, 64, 32};
    _Float16* cB[6];
    for (int i = 0; i < 6; ++i) cB[i] = aH((long long)R2_ * cS[i]);
    _Float16* u1 = aH((long long)R2_ * 256);

    float* uv1 = aF((long long)R3_ * 2);
    float* uv2 = aF((long long)R3_ * 2);
    float* m1v = aF(R3_);
    float* m2v = aF(R3_);
    float* accB[4]; float* cntB[4];
    for (int i = 0; i < 4; ++i) { accB[i] = aF((long long)R2_ * 64); cntB[i] = aF(R2_); }
    int* idx1 = aI(RKr_);
    int* idx2 = aI(RKr_);
    float* fd2  = aF((long long)R2_ * 2);
    float* v1b  = aF((long long)R2_ * 2);
    float* fcf  = aF((long long)R2_ * 2);
    float* fd3  = aF((long long)R3_ * 3);
    float* umO  = aF((long long)R2_ * 144);

    if (off > ws_size) return;   // not enough scratch; nothing we can do

    // zero everything: clears channel pads, weight pads, zero-row, atomics
    hipMemsetAsync(d_ws, 0, off, stream);

    // weight conversion fp32 -> padded f16
    for (int i = 0; i < nws; ++i) {
        long long tot = (long long)WS[i].Co * WS[i].Ci * WS[i].taps;
        k_wcvt<<<g1(tot), 256, 0, stream>>>(WS[i].w, WS[i].wf, WS[i].Co, WS[i].Ci,
                                            WS[i].taps, WS[i].Kpad, WS[i].sm);
        k_copyf<<<g1(WS[i].Co), 256, 0, stream>>>(WS[i].b, WS[i].bf, WS[i].Co);
    }

    auto G = [&](int wi, const _Float16* in, int inStride, int rows, int dil,
                 _Float16* outR, int outStride, int outChOff,
                 float* out32, int o32Stride, int act) {
        const WSpec& q = WS[wi];
        int waves = (q.CoPad >> 5) * (rows >> 5);
        k_wmma_conv<<<dim3((waves + 7) / 8), dim3(256), 0, stream>>>(
            in, q.wf, q.bf, zrow, outR, out32, rows, q.Kpad, q.taps, dil,
            H_, W_, inStride, q.CoPad, q.Co, outStride, outChOff, o32Stride, act);
    };

    // 1) inputs -> f16 rows
    k_rows_from_nchw<<<g1((long long)R2_ * 64), 256, 0, stream>>>(feat1_2d, cat2d1, 64, HW_, 128, 0, B_);
    k_rows_from_nchw<<<g1((long long)R2_ * 64), 256, 0, stream>>>(feat2_2d, cat2d2, 64, HW_, 128, 0, B_);
    k_rows_from_nchw<<<g1((long long)R3_ * 64), 256, 0, stream>>>(feat1_3d, cat3d1, 64, N_, 128, 0, B_);
    k_rows_from_nchw<<<g1((long long)R3_ * 64), 256, 0, stream>>>(feat2_3d, cat3d2, 64, N_, 128, 0, B_);

    // 2) projection + knn
    k_project<<<g1(R3_), 256, 0, stream>>>(xyz1, cam, uv1, m1v);
    k_project<<<g1(R3_), 256, 0, stream>>>(xyz2, cam, uv2, m2v);
    k_knn<<<g1(R3_), 256, 0, stream>>>(xyz1, nullptr, xyz1, idx1);
    k_knn<<<g1(R3_), 256, 0, stream>>>(xyz1, lf3d, xyz2, idx2);

    // 3) fuse frame 1 and frame 2
    k_gather<<<g1((long long)R3_ * 64), 256, 0, stream>>>(cat2d1, 128, 0, 64, uv1, m1v, cat3d1, 128, 64);
    G(iWp3d, cat3d1, 128, R3_, 1, f1_3d, 64, 0, nullptr, 0, 1);
    k_scatter<<<g1((long long)R3_ * 64), 256, 0, stream>>>(cat3d1, 128, 0, 64, uv1, m1v, accB[0], cntB[0]);
    k_scatnorm<<<g1((long long)R2_ * 64), 256, 0, stream>>>(accB[0], cntB[0], 64, cat2d1, 128, 64);
    G(iWp2d, cat2d1, 128, R2_, 1, f1_2d, 64, 0, nullptr, 0, 1);

    k_gather<<<g1((long long)R3_ * 64), 256, 0, stream>>>(cat2d2, 128, 0, 64, uv2, m2v, cat3d2, 128, 64);
    G(iWp3d, cat3d2, 128, R3_, 1, f2_3d, 64, 0, nullptr, 0, 1);
    k_scatter<<<g1((long long)R3_ * 64), 256, 0, stream>>>(cat3d2, 128, 0, 64, uv2, m2v, accB[1], cntB[1]);
    k_scatnorm<<<g1((long long)R2_ * 64), 256, 0, stream>>>(accB[1], cntB[1], 64, cat2d2, 128, 64);
    G(iWp2d, cat2d2, 128, R2_, 1, f2_2d, 64, 0, nullptr, 0, 1);

    // 4) backwarp + 2D correlation
    k_backwarp<<<g1((long long)R2_ * 64), 256, 0, stream>>>(f2_2d, 64, lf2d, warp2, 64);
    k_corr<<<g1((long long)R2_ * CORR_), 256, 0, stream>>>(f1_2d, 64, warp2, 64, corr2c, 160);

    // 5) 3D correlation branch
    k_cat1<<<g1((long long)RKr_ * 131), 256, 0, stream>>>(f1_3d, f2_3d, xyz1, lf3d, xyz2, idx2, cat1b);
    G(iWc1, cat1b, 160, RKr_, 1, hbuf, 64, 0, nullptr, 0, 1);
    k_maxk<<<g1((long long)R3_ * 64), 256, 0, stream>>>(hbuf, 64, 64, hmax, 64, 0);
    k_cat2<<<g1((long long)RKr_ * 67), 256, 0, stream>>>(hmax, xyz1, idx1, cat2b);
    G(iWc2, cat2b, 96, RKr_, 1, h2buf, 64, 0, nullptr, 0, 1);
    k_maxk<<<g1((long long)R3_ * 64), 256, 0, stream>>>(h2buf, 64, 64, corr3c, 160, 0);

    // 6) correlation fuse
    k_gather<<<g1((long long)R3_ * 81), 256, 0, stream>>>(corr2c, 160, 0, 81, uv1, m1v, corr3c, 160, 64);
    G(iWf3d, corr3c, 160, R3_, 1, corr3f, 64, 0, nullptr, 0, 1);
    k_scatter<<<g1((long long)R3_ * 64), 256, 0, stream>>>(corr3c, 160, 0, 64, uv1, m1v, accB[2], cntB[2]);
    k_scatnorm<<<g1((long long)R2_ * 64), 256, 0, stream>>>(accB[2], cntB[2], 64, corr2c, 160, 81);
    G(iWf2d, corr2c, 160, R2_, 1, estB, 704, 0, nullptr, 0, 1);   // x2d[0..80]

    // 7) x2d assembly
    G(iWa2d, f1_2d, 64, R2_, 1, estB, 704, 81, nullptr, 0, 1);    // a2d -> [81..144]
    k_rows_from_nchw<<<g1((long long)R2_ * 2), 256, 0, stream>>>(lf2d, estB, 2, HW_, 704, 145, B_);
    k_rows_from_nchw<<<g1((long long)R2_ * 32), 256, 0, stream>>>(lff2d, estB, 32, HW_, 704, 147, B_);

    // 8) estimator chain (dense growing concat, 32-aligned segments)
    for (int i = 0; i < 5; ++i)
        G(iWe[i], estB, 704, R2_, 1, estB, 704, estKp[i], nullptr, 0, 1);

    // 9) 3D flow-feature branch
    G(iWac3d, corr3f, 64, R3_, 1, x3dB, 224, 0, nullptr, 0, 1);
    G(iWa3d, f1_3d, 64, R3_, 1, x3dB, 224, 64, nullptr, 0, 1);
    k_rows_from_nchw<<<g1((long long)R3_ * 3), 256, 0, stream>>>(lf3d, x3dB, 3, N_, 224, 128, B_);
    k_rows_from_nchw<<<g1((long long)R3_ * 64), 256, 0, stream>>>(lff3d, x3dB, 64, N_, 224, 131, B_);
    G(iWfp1, x3dB, 224, R3_, 1, h3a, 128, 0, nullptr, 0, 1);
    k_maxgroup<<<g1((long long)R3_ * 128), 256, 0, stream>>>(h3a, 128, 128, idx1, h3b, 128);
    G(iWfp2, h3b, 128, R3_, 1, h3c, 128, 0, nullptr, 0, 1);
    G(iWfp3, h3c, 128, R3_, 1, cat33, 704, 0, nullptr, 0, 1);     // ff3d

    // 10) final fuse
    k_gather<<<g1((long long)R3_ * 640), 256, 0, stream>>>(estB, 704, 0, 640, uv1, m1v, cat33, 704, 64);
    G(iWef3d, cat33, 704, R3_, 1, ff3dn, 64, 0, nullptr, 0, 1);
    k_scatter<<<g1((long long)R3_ * 64), 256, 0, stream>>>(cat33, 704, 0, 64, uv1, m1v, accB[3], cntB[3]);
    k_scatnorm<<<g1((long long)R2_ * 64), 256, 0, stream>>>(accB[3], cntB[3], 64, estB, 704, 640);
    G(iWef2d, estB, 704, R2_, 1, hcB, 640, 0, nullptr, 0, 1);     // new ff2d (dense 627)

    // 11) flow heads
    G(iWl3d, ff3dn, 64, R3_, 1, nullptr, 0, 0, fd3, 3, 0);
    k_flow3d_final<<<g1(R3_ * 3), 256, 0, stream>>>(lf3d, fd3, outFlow3);

    G(iWl2d, hcB, 640, R2_, 1, nullptr, 0, 0, fd2, 2, 0);
    k_flow2d_mid<<<g1(R2_ * 2), 256, 0, stream>>>(lf2d, fd2, v1b, hcB, 640, 627);

    // 12) context network (dilated convs)
    const int dils[6] = {1, 2, 4, 8, 16, 1};
    const _Float16* cin = hcB; int cis = 640;
    for (int i = 0; i < 6; ++i) {
        G(iWctx[i], cin, cis, R2_, dils[i], cB[i], cS[i], 0, nullptr, 0, 1);
        cin = cB[i]; cis = cS[i];
    }
    G(iWcf, cB[5], 32, R2_, 1, nullptr, 0, 0, fcf, 2, 0);
    k_flow2d_final<<<g1(R2_ * 2), 256, 0, stream>>>(v1b, fcf, outFlow2);

    // 13) up-mask head
    G(iWum1, cB[5], 32, R2_, 1, u1, 256, 0, nullptr, 0, 2);
    G(iWum2, u1, 256, R2_, 1, nullptr, 0, 0, umO, 144, 0);
    k_rows_to_nchw<<<g1((long long)R2_ * 144), 256, 0, stream>>>(umO, 144, outUm);
}
